// GraphEncoder_87058987090026
// MI455X (gfx1250) — matrix operand order repaired and checked by reading, not compiled, not run on hardware
//
#include <hip/hip_runtime.h>
#include <hip/hip_bf16.h>

// ---------------------------------------------------------------------------
// GraphSAGE-style 3-layer GNN, B=8192 molecules x A=32 atoms x H=128 features.
// bf16 WMMA (v_wmma_f32_16x16x32_bf16) with f32 accumulation.
// One 256-thread block (8 wave32) per molecule; wave w owns N-columns [16w,16w+16).
// ---------------------------------------------------------------------------

typedef __attribute__((ext_vector_type(16))) __bf16 v16bf;
typedef __attribute__((ext_vector_type(8)))  float  v8f;

union FragBF {            // 16 bf16 elements per lane = A or B operand
    v16bf v;
    uint4 q[2];
};
union FragF32 {           // 8 f32 elements per lane = C/D accumulator
    v8f   v;
    float f[8];
};

// native HW bf16 convert (v_cvt_pk_bf16_f32 class), two floats -> packed u32
__device__ __forceinline__ unsigned pk2bf(float a, float b) {
    __bf16 lo = (__bf16)a;
    __bf16 hi = (__bf16)b;
    return (unsigned)__builtin_bit_cast(unsigned short, lo) |
           ((unsigned)__builtin_bit_cast(unsigned short, hi) << 16);
}
__device__ __forceinline__ unsigned short f2bf(float f) {
    return __builtin_bit_cast(unsigned short, (__bf16)f);
}

#define WMMA_BF16(A_, B_, C_) \
    __builtin_amdgcn_wmma_f32_16x16x32_bf16(false, (A_).v, false, (B_).v, (short)0, (C_).v, false, false)

static constexpr int NA = 32;    // atoms per molecule
static constexpr int HH = 128;   // hidden dim

// --- prep: convert the six [128][128] fp32 weight matrices to bf16 in d_ws ---
__global__ void cvt_weights_bf16(const float* __restrict__ wl1, const float* __restrict__ wr1,
                                 const float* __restrict__ wl2, const float* __restrict__ wr2,
                                 const float* __restrict__ wl3, const float* __restrict__ wr3,
                                 unsigned* __restrict__ ws) {           // packed pairs
    int gid = blockIdx.x * blockDim.x + threadIdx.x;                    // one pair each
    if (gid >= 6 * HH * HH / 2) return;
    const float* srcs[6] = {wl1, wr1, wl2, wr2, wl3, wr3};
    int m = gid / (HH * HH / 2);
    int i = (gid - m * (HH * HH / 2)) * 2;
    const float2 v = *(const float2*)&srcs[m][i];
    ws[gid] = pk2bf(v.x, v.y);
}

// --- main kernel: one block per molecule ---
__global__ __launch_bounds__(256) void gnn_sage_wmma(
        const int*   __restrict__ z,
        const float* __restrict__ pos,
        const float* __restrict__ embed,
        const unsigned short* __restrict__ ws,   // [wl1|wr1|wl2|wr2|wl3|wr3] bf16, each 128*128
        const float* __restrict__ b1,
        const float* __restrict__ b2,
        const float* __restrict__ b3,
        float* __restrict__ out) {
    __shared__ __align__(16) unsigned short sX [NA * HH];   // x, row-major   [atom][feat] bf16
    __shared__ __align__(16) unsigned short sXT[HH * NA];   // x, feat-major  [feat][atom] bf16
    __shared__ __align__(16) unsigned short sN [NA * HH];   // neigh, row-major bf16
    __shared__ __align__(16) unsigned short sAdj[NA * NA];  // adjacency bf16 (0/1)
    __shared__ float sPos[NA * 3];
    __shared__ float sDegInv[NA];
    __shared__ int   sZ[NA];

    const int mol   = blockIdx.x;
    const int tid   = threadIdx.x;
    const int lane  = tid & 31;
    const int wv    = tid >> 5;          // wave id 0..7
    const int nBase = wv * 16;           // this wave's N columns
    const int hi    = (lane >= 16) ? 1 : 0;
    const int nCol  = lane & 15;         // N (and A-row) index within tile
    const int col   = nBase + nCol;      // this lane's feature column

    // ---- stage positions + atom types ----
    if (tid < NA * 3) sPos[tid] = pos[mol * (NA * 3) + tid];
    if (tid < NA)     sZ[tid]   = z[mol * NA + tid];
    __syncthreads();

    // ---- adjacency + degree (threads 0..31, one row each, packed stores) ----
    if (tid < NA) {
        const float xi = sPos[tid * 3 + 0];
        const float yi = sPos[tid * 3 + 1];
        const float zi = sPos[tid * 3 + 2];
        float dsum = 0.0f;
        for (int j0 = 0; j0 < NA; j0 += 8) {
            unsigned p[4];
#pragma unroll
            for (int u = 0; u < 4; ++u) {
                unsigned pk = 0;
#pragma unroll
                for (int v = 0; v < 2; ++v) {
                    int j = j0 + 2 * u + v;
                    float dx = xi - sPos[j * 3 + 0];
                    float dy = yi - sPos[j * 3 + 1];
                    float dz = zi - sPos[j * 3 + 2];
                    float d2 = dx * dx + dy * dy + dz * dz;
                    bool  e  = (d2 <= 25.0f) && (j != tid);
                    pk |= (e ? 0x3F80u : 0u) << (16 * v);    // bf16 1.0/0.0
                    dsum += e ? 1.0f : 0.0f;
                }
                p[u] = pk;
            }
            *(uint4*)&sAdj[tid * NA + j0] = make_uint4(p[0], p[1], p[2], p[3]);
        }
        sDegInv[tid] = 1.0f / fmaxf(dsum, 1.0f);
    }
    __syncthreads();   // sZ ready for everyone below (and adj/deg for layer loop)

    // ---- x = embed[z]: vectorized row-major writes + transpose scatter ----
    for (int e = tid; e < NA * HH / 8; e += 256) {       // 512 groups of 8 feats
        int atom = e >> 4;
        int h8   = (e & 15) * 8;
        const float4 v0 = *(const float4*)&embed[sZ[atom] * HH + h8];
        const float4 v1 = *(const float4*)&embed[sZ[atom] * HH + h8 + 4];
        uint4 p;
        p.x = pk2bf(v0.x, v0.y);  p.y = pk2bf(v0.z, v0.w);
        p.z = pk2bf(v1.x, v1.y);  p.w = pk2bf(v1.z, v1.w);
        *(uint4*)&sX[atom * HH + h8] = p;
        const unsigned short* hp = (const unsigned short*)&p;
#pragma unroll
        for (int i = 0; i < 8; ++i)
            sXT[(h8 + i) * NA + atom] = hp[i];
    }
    __syncthreads();

    // per-lane 1/deg for the 16 rows this lane produces (C-layout: M = r + 8*hi + 16*mt)
    float dInv[2][8];
#pragma unroll
    for (int mt = 0; mt < 2; ++mt)
#pragma unroll
        for (int r = 0; r < 8; ++r)
            dInv[mt][r] = sDegInv[mt * 16 + r + hi * 8];

    const float* bptr[3] = {b1, b2, b3};

#pragma unroll
    for (int layer = 0; layer < 3; ++layer) {
        const unsigned short* Wl = ws + (2 * layer + 0) * HH * HH;
        const unsigned short* Wr = ws + (2 * layer + 1) * HH * HH;
        __builtin_prefetch(Wl + col * HH, 0, 0);
        __builtin_prefetch(Wr + col * HH, 0, 0);
        const float bval = bptr[layer][col];

        // ===== Phase A: neigh = (adj @ x) * (1/deg)  (K = 32, one WMMA per M-tile)
        {
            FragBF bx;   // B-frag from xT: col n, runs K+0..7 and K+16..23 per half-wave
            const uint4* pb = (const uint4*)&sXT[col * NA + hi * 8];
            bx.q[0] = pb[0]; bx.q[1] = pb[2];

            FragBF a0, a1;  // A-frags from adj rows (runs K+0..7 and K+16..23 per half-wave)
            const uint4* p0 = (const uint4*)&sAdj[(0  + nCol) * NA + hi * 8];
            a0.q[0] = p0[0]; a0.q[1] = p0[2];
            const uint4* p1 = (const uint4*)&sAdj[(16 + nCol) * NA + hi * 8];
            a1.q[0] = p1[0]; a1.q[1] = p1[2];

            FragF32 c0, c1;
#pragma unroll
            for (int r = 0; r < 8; ++r) { c0.f[r] = 0.0f; c1.f[r] = 0.0f; }
            c0.v = WMMA_BF16(a0, bx, c0);
            c1.v = WMMA_BF16(a1, bx, c1);

#pragma unroll
            for (int r = 0; r < 8; ++r) {
                sN[(0  + r + hi * 8) * HH + col] = f2bf(c0.f[r] * dInv[0][r]);
                sN[(16 + r + hi * 8) * HH + col] = f2bf(c1.f[r] * dInv[1][r]);
            }
        }
        __syncthreads();

        // ===== Phase B: y = relu(neigh @ Wl^T + b + x @ Wr^T)  (K = 128, 4 chunks)
        // hoist this wave's weight B-frags: B(k,n) = W[n*128 + k] -> contiguous in k
        FragBF wlF[4], wrF[4];
#pragma unroll
        for (int kc = 0; kc < 4; ++kc) {
            const uint4* pl = (const uint4*)&Wl[col * HH + kc * 32 + hi * 8];
            wlF[kc].q[0] = pl[0]; wlF[kc].q[1] = pl[2];
            const uint4* pr = (const uint4*)&Wr[col * HH + kc * 32 + hi * 8];
            wrF[kc].q[0] = pr[0]; wrF[kc].q[1] = pr[2];
        }

        FragF32 cc0, cc1;
#pragma unroll
        for (int r = 0; r < 8; ++r) { cc0.f[r] = bval; cc1.f[r] = bval; }

#pragma unroll
        for (int kc = 0; kc < 4; ++kc) {
            FragBF an0, an1, ax0, ax1;
            const uint4* pn0 = (const uint4*)&sN[(0  + nCol) * HH + kc * 32 + hi * 8];
            an0.q[0] = pn0[0]; an0.q[1] = pn0[2];
            const uint4* pn1 = (const uint4*)&sN[(16 + nCol) * HH + kc * 32 + hi * 8];
            an1.q[0] = pn1[0]; an1.q[1] = pn1[2];
            const uint4* px0 = (const uint4*)&sX[(0  + nCol) * HH + kc * 32 + hi * 8];
            ax0.q[0] = px0[0]; ax0.q[1] = px0[2];
            const uint4* px1 = (const uint4*)&sX[(16 + nCol) * HH + kc * 32 + hi * 8];
            ax1.q[0] = px1[0]; ax1.q[1] = px1[2];

            // interleave the two independent accumulator chains to hide WMMA latency
            cc0.v = WMMA_BF16(an0, wlF[kc], cc0);
            cc1.v = WMMA_BF16(an1, wlF[kc], cc1);
            cc0.v = WMMA_BF16(ax0, wrF[kc], cc0);
            cc1.v = WMMA_BF16(ax1, wrF[kc], cc1);
        }
#pragma unroll
        for (int r = 0; r < 8; ++r) {
            cc0.f[r] = fmaxf(cc0.f[r], 0.0f);
            cc1.f[r] = fmaxf(cc1.f[r], 0.0f);
        }

        if (layer < 2) {
            __syncthreads();  // everyone done reading old x/xT

            // packed converts (pairs -> v_cvt_pk_bf16_f32 class)
            uint4 pk0, pk1;
            pk0.x = pk2bf(cc0.f[0], cc0.f[1]);  pk0.y = pk2bf(cc0.f[2], cc0.f[3]);
            pk0.z = pk2bf(cc0.f[4], cc0.f[5]);  pk0.w = pk2bf(cc0.f[6], cc0.f[7]);
            pk1.x = pk2bf(cc1.f[0], cc1.f[1]);  pk1.y = pk2bf(cc1.f[2], cc1.f[3]);
            pk1.z = pk2bf(cc1.f[4], cc1.f[5]);  pk1.w = pk2bf(cc1.f[6], cc1.f[7]);

            // sXT writeback is contiguous in M for r=0..7: one b128 store per tile
            *(uint4*)&sXT[col * NA + 0  + hi * 8] = pk0;
            *(uint4*)&sXT[col * NA + 16 + hi * 8] = pk1;

            // sX row-major writeback (inherent transpose scatter)
            const unsigned short* h0 = (const unsigned short*)&pk0;
            const unsigned short* h1 = (const unsigned short*)&pk1;
#pragma unroll
            for (int r = 0; r < 8; ++r) {
                sX[(0  + r + hi * 8) * HH + col] = h0[r];
                sX[(16 + r + hi * 8) * HH + col] = h1[r];
            }
            __syncthreads();
        } else {
            // ===== global_add_pool: sum over 32 atoms, straight from registers.
            // Lane L (<16) holds rows {0-7,16-23}, lane L+16 holds {8-15,24-31}, same column.
            float s = 0.0f;
#pragma unroll
            for (int r = 0; r < 8; ++r) s += cc0.f[r] + cc1.f[r];
            s += __shfl_xor(s, 16, 32);
            if (lane < 16) out[mol * HH + col] = s;
        }
    }
}

extern "C" void kernel_launch(void* const* d_in, const int* in_sizes, int n_in,
                              void* d_out, int out_size, void* d_ws, size_t ws_size,
                              hipStream_t stream) {
    const int*   z     = (const int*)  d_in[0];
    const float* pos   = (const float*)d_in[1];
    const float* embed = (const float*)d_in[2];
    const float* Wl1   = (const float*)d_in[3];
    const float* bl1   = (const float*)d_in[4];
    const float* Wr1   = (const float*)d_in[5];
    const float* Wl2   = (const float*)d_in[6];
    const float* bl2   = (const float*)d_in[7];
    const float* Wr2   = (const float*)d_in[8];
    const float* Wl3   = (const float*)d_in[9];
    const float* bl3   = (const float*)d_in[10];
    const float* Wr3   = (const float*)d_in[11];

    unsigned* wbf = (unsigned*)d_ws;               // 6*128*128 bf16 = 192 KB

    const int nMol = in_sizes[0] / NA;             // 8192

    cvt_weights_bf16<<<(6 * HH * HH / 2 + 255) / 256, 256, 0, stream>>>(
        Wl1, Wr1, Wl2, Wr2, Wl3, Wr3, wbf);
    gnn_sage_wmma<<<nMol, 256, 0, stream>>>(
        z, pos, embed, (const unsigned short*)d_ws, bl1, bl2, bl3, (float*)d_out);
}
